// NCELoss_70978629534242
// MI455X (gfx1250) — matrix operand order repaired and checked by reading, not compiled, not run on hardware
//
#include <hip/hip_runtime.h>
#include <hip/hip_bf16.h>

typedef __attribute__((ext_vector_type(16))) _Float16 v16h;
typedef __attribute__((ext_vector_type(8)))  float    v8f;
typedef __attribute__((ext_vector_type(4)))  float    v4f;

#define EMB       1024
#define KNEG      10
#define NORM_TERM 9.0f

// One wave (32 lanes) per token. A = 16 gathered weight rows (f32->f16),
// B = input row chunk broadcast to all 16 columns. 32 x v_wmma_f32_16x16x32_f16
// accumulates the 16 logits (rows 11..15 are duplicates of row 10 -> their
// loads alias row 10's cache lines; masked out of the loss).
__global__ __launch_bounds__(256) void nce_wmma_kernel(
    const float* __restrict__ input,    // [N, 1024]
    const float* __restrict__ weight,   // [50257, 1024]
    const float* __restrict__ bias,     // [50257]
    const float* __restrict__ noise,    // [50257]
    const int*   __restrict__ target,   // [N]
    const int*   __restrict__ ns,       // [N, 10]
    float*       __restrict__ per_token,
    int N)
{
    const int lane = threadIdx.x & 31;
    const int wave = threadIdx.x >> 5;
    const int n    = blockIdx.x * 8 + wave;
    if (n >= N) return;                      // wave-uniform: EXEC stays all-1s

    const int m = lane & 15;                 // matrix row this lane feeds
    const int g = lane >> 4;                 // lane group (K-half selector)

    int row;
    if (m == 0)        row = target[n];
    else if (m <= KNEG) row = ns[n * KNEG + (m - 1)];
    else                row = ns[n * KNEG + (KNEG - 1)];   // duplicate -> free loads

    const float* wRow = weight + (long)row * EMB;
    const float* xRow = input  + (long)n   * EMB;

    v8f c = {};
    #pragma unroll 2
    for (int s = 0; s < EMB / 32; ++s) {
        const int k0 = s * 32;
        // A-matrix (16-bit 16x32 layout): group0 lanes: K = k0+[0..7], k0+16+[0..7]
        //                                 group1 lanes: K = k0+8+[0..7], k0+24+[0..7]
        v4f wa0 = *(const v4f*)(wRow + k0 + 8 * g);
        v4f wa1 = *(const v4f*)(wRow + k0 + 8 * g + 4);
        v4f wa2 = *(const v4f*)(wRow + k0 + 16 + 8 * g);
        v4f wa3 = *(const v4f*)(wRow + k0 + 16 + 8 * g + 4);
        // B-matrix (16-bit 32x16 layout): group0 lanes: K = k0+[0..7], k0+16+[0..7]
        //                                 group1 lanes: K = k0+8+[0..7], k0+24+[0..7]
        // Same values across all lanes of a group -> every C column = logits.
        v4f xb0 = *(const v4f*)(xRow + k0 + 8 * g);
        v4f xb1 = *(const v4f*)(xRow + k0 + 8 * g + 4);
        v4f xb2 = *(const v4f*)(xRow + k0 + 16 + 8 * g);
        v4f xb3 = *(const v4f*)(xRow + k0 + 16 + 8 * g + 4);

        v16h a, b;
        #pragma unroll
        for (int i = 0; i < 4; ++i) {
            a[i]      = (_Float16)wa0[i];
            a[4 + i]  = (_Float16)wa1[i];
            a[8 + i]  = (_Float16)wa2[i];
            a[12 + i] = (_Float16)wa3[i];
            b[i]      = (_Float16)xb0[i];
            b[4 + i]  = (_Float16)xb1[i];
            b[8 + i]  = (_Float16)xb2[i];
            b[12 + i] = (_Float16)xb3[i];
        }
        c = __builtin_amdgcn_wmma_f32_16x16x32_f16(
                /*neg_a=*/false, a, /*neg_b=*/false, b,
                /*c_mod=*/(short)0, c, /*reuse_a=*/false, /*reuse_b=*/false);
    }

    // C layout: lane (group g) holds C[M = 8g + v, N = m] in c[v], all columns
    // identical. Each lane evaluates the loss terms for its 8 rows.
    float partial = 0.0f;
    #pragma unroll
    for (int j = 0; j < 8; ++j) {
        const int M = 8 * g + j;
        const int rj = __shfl(row, M, 32);        // lane M (<16) holds row idx M
        if (M <= KNEG) {
            const float logit = c[j] + bias[rj];
            const float prob  = __expf(logit - NORM_TERM);
            const float kn    = (float)KNEG * noise[rj];
            if (M == 0) partial += __logf(prob / (prob + kn));  // rnn_loss
            else        partial += __logf(kn   / (prob + kn));  // noise_loss
        }
    }
    // group0 holds terms M=0..7, group1 holds M=8..10; combine across groups.
    const float tok = partial + __shfl_xor(partial, 16, 32);
    if (lane == 0) per_token[n] = tok;
}

// Deterministic single-block tree reduction: loss = -sum(per_token)/N.
__global__ __launch_bounds__(256) void nce_reduce_kernel(
    const float* __restrict__ per_token, float* __restrict__ out, int N)
{
    __shared__ float buf[256];
    float s = 0.0f;
    for (int i = threadIdx.x; i < N; i += 256) s += per_token[i];
    buf[threadIdx.x] = s;
    __syncthreads();
    for (int off = 128; off > 0; off >>= 1) {
        if ((int)threadIdx.x < off) buf[threadIdx.x] += buf[threadIdx.x + off];
        __syncthreads();
    }
    if (threadIdx.x == 0) out[0] = -buf[0] / (float)N;
}

extern "C" void kernel_launch(void* const* d_in, const int* in_sizes, int n_in,
                              void* d_out, int out_size, void* d_ws, size_t ws_size,
                              hipStream_t stream) {
    const float* input  = (const float*)d_in[0];
    const float* weight = (const float*)d_in[1];
    const float* bias   = (const float*)d_in[2];
    const float* noise  = (const float*)d_in[3];
    const int*   target = (const int*)d_in[4];
    const int*   nsamp  = (const int*)d_in[5];
    const int N = in_sizes[4];

    float* per_token = (float*)d_ws;   // N floats of scratch
    float* out       = (float*)d_out;  // scalar loss

    const int blocks = (N + 7) / 8;    // 8 waves (tokens) per 256-thread block
    nce_wmma_kernel<<<blocks, 256, 0, stream>>>(input, weight, bias, noise,
                                                target, nsamp, per_token, N);
    nce_reduce_kernel<<<1, 256, 0, stream>>>(per_token, out, N);
}
